// GlobalBPCAPooling2D_90683939488319
// MI455X (gfx1250) — matrix operand order repaired and checked by reading, not compiled, not run on hardware
//
#include <hip/hip_runtime.h>
#include <hip/hip_bf16.h>
#include <math.h>

typedef __attribute__((ext_vector_type(16))) _Float16 v16h;
typedef __attribute__((ext_vector_type(8)))  _Float16 v8h;
typedef __attribute__((ext_vector_type(8)))  float    v8f;

#define BATCH 128
#define ROWS  768
#define COLS  256
#define NITER 256
#define TSTRIDE 40   // f16 elements per LDS column: 80B, 16B-aligned, decent bank spread

// ---------------------------------------------------------------------------
// Kernel 1: per-column mean and 1/std (population std, matching jnp.std).
// invstd = 0 when var == 0 -> standardized column becomes 0 (matches NaN->0).
// ---------------------------------------------------------------------------
__global__ __launch_bounds__(COLS) void k_stats(const float* __restrict__ x,
                                                float* __restrict__ mean,
                                                float* __restrict__ invstd) {
  int b = blockIdx.x, c = threadIdx.x;
  const float* xb = x + (size_t)b * ROWS * COLS;
  float s = 0.f, s2 = 0.f;
#pragma unroll 8
  for (int r = 0; r < ROWS; ++r) {
    float v = xb[(size_t)r * COLS + c];
    s += v; s2 += v * v;
  }
  float m   = s * (1.0f / ROWS);
  float var = fmaxf(s2 * (1.0f / ROWS) - m * m, 0.0f);
  mean[b * COLS + c]   = m;
  invstd[b * COLS + c] = (var > 0.f) ? rsqrtf(var) : 0.f;
}

// ---------------------------------------------------------------------------
// Kernel 2: G = P^T P via v_wmma_f32_16x16x32_f16.
// grid = (4 j-strips, 128 batches); block = 256 threads = 8 waves.
// Wave w owns i-tiles {2w, 2w+1}; block strip owns 4 j-tiles -> 8 acc tiles/wave.
// K=768 streamed as 24 chunks of 32 rows staged column-major in LDS.
// ---------------------------------------------------------------------------
__global__ __launch_bounds__(256) void k_gram(const float* __restrict__ x,
                                              const float* __restrict__ mean,
                                              const float* __restrict__ invstd,
                                              float* __restrict__ G) {
  __shared__ __align__(16) _Float16 tile[COLS * TSTRIDE];  // 20 KB

  int b     = blockIdx.y;
  int jbase = blockIdx.x * 64;
  int t     = threadIdx.x;
  int lane  = t & 31;
  int w     = t >> 5;

  const float* xb = x + (size_t)b * ROWS * COLS;
  float m  = mean[b * COLS + t];
  float is = invstd[b * COLS + t];
  float* Gb = G + (size_t)b * COLS * COLS;

  v8f zero{};
  v8f acc[8];
#pragma unroll
  for (int i = 0; i < 8; ++i) acc[i] = zero;

  int i0 = w * 32;                 // this wave's two i-tiles: i0, i0+16
  int cc = lane & 15;
  int khA = (lane >> 4) << 3;      // A frag: 0 or 8
  int khB = (lane >> 4) << 3;      // B frag: 0 or 8

  for (int ch = 0; ch < ROWS / 32; ++ch) {
    __syncthreads();               // previous chunk fully consumed
    int k0 = ch * 32;
    // stage standardized f16 chunk, transposed: tile[col][k_local]
#pragma unroll
    for (int kl = 0; kl < 32; ++kl)
      tile[t * TSTRIDE + kl] = (_Float16)((xb[(size_t)(k0 + kl) * COLS + t] - m) * is);
    __syncthreads();

    v16h a[2], bb[4];
    // A fragments (16x32 f16): lanes 0-15 -> k {0..7,16..23}, lanes 16-31 -> k {8..15,24..31}
#pragma unroll
    for (int ii = 0; ii < 2; ++ii) {
      const _Float16* p = &tile[(i0 + ii * 16 + cc) * TSTRIDE];
      v8h lo = *(const v8h*)(p + khA);
      v8h hi = *(const v8h*)(p + 16 + khA);
      v16h f;
#pragma unroll
      for (int e = 0; e < 8; ++e) { f[e] = lo[e]; f[8 + e] = hi[e]; }
      a[ii] = f;
    }
    // B fragments (32x16 f16): lanes 0-15 -> k {0..7,16..23}, lanes 16-31 -> k {8..15,24..31}
#pragma unroll
    for (int jj = 0; jj < 4; ++jj) {
      const _Float16* p = &tile[(jbase + jj * 16 + cc) * TSTRIDE];
      v8h lo = *(const v8h*)(p + khB);
      v8h hi = *(const v8h*)(p + 16 + khB);
      v16h f;
#pragma unroll
      for (int e = 0; e < 8; ++e) { f[e] = lo[e]; f[8 + e] = hi[e]; }
      bb[jj] = f;
    }
#pragma unroll
    for (int ii = 0; ii < 2; ++ii)
#pragma unroll
      for (int jj = 0; jj < 4; ++jj)
        acc[ii * 4 + jj] = __builtin_amdgcn_wmma_f32_16x16x32_f16(
            false, a[ii], false, bb[jj], (short)0, acc[ii * 4 + jj], false, false);
  }

  // C/D layout: VGPR e -> row M = e + 8*(lane>>4), col N = lane&15
  int colN = lane & 15;
  int rsel = (lane >> 4) * 8;
#pragma unroll
  for (int ii = 0; ii < 2; ++ii) {
    int gi0 = i0 + ii * 16;
#pragma unroll
    for (int jj = 0; jj < 4; ++jj) {
      int gj = jbase + jj * 16 + colN;
      v8f c = acc[ii * 4 + jj];
#pragma unroll
      for (int e = 0; e < 8; ++e)
        Gb[(size_t)(gi0 + e + rsel) * COLS + gj] = c[e];
    }
  }
}

// ---------------------------------------------------------------------------
// Kernel 3: power iteration on G (in LDS, padded rows) + projection out = P v.
// grid = 128 batches, block = 256 threads. Dynamic LDS = 256*257*4 = 257 KB.
// ---------------------------------------------------------------------------
__global__ __launch_bounds__(COLS) void k_power_proj(const float* __restrict__ x,
                                                     const float* __restrict__ mean,
                                                     const float* __restrict__ invstd,
                                                     const float* __restrict__ G,
                                                     float* __restrict__ out) {
  extern __shared__ float Gs[];           // [256][257] padded -> conflict-free rows
  __shared__ float vv[COLS];
  __shared__ float wvec[COLS];
  __shared__ float part[8];
  __shared__ float sgn_sh;

  int b = blockIdx.x, t = threadIdx.x;
  int lane = t & 31, wid = t >> 5;
  const float* Gb = G + (size_t)b * COLS * COLS;

  for (int idx = t; idx < COLS * COLS; idx += COLS) {
    int r = idx >> 8, c = idx & 255;
    Gs[r * 257 + c] = Gb[idx];
  }
  vv[t] = 1.0f + 0.0009765625f * (float)t;   // deterministic, not orthogonal to v1
  __syncthreads();

  for (int it = 0; it < NITER; ++it) {
    float y = 0.f;
    const float* row = &Gs[t * 257];
#pragma unroll 4
    for (int k = 0; k < COLS; ++k) y = fmaf(row[k], vv[k], y);
    float s = y * y;
#pragma unroll
    for (int off = 16; off; off >>= 1) s += __shfl_xor(s, off, 32);
    __syncthreads();                         // all vv reads done
    if (lane == 0) part[wid] = s;
    __syncthreads();
    float nrm = part[0] + part[1] + part[2] + part[3] +
                part[4] + part[5] + part[6] + part[7];
    float inv = (nrm > 0.f) ? rsqrtf(nrm) : 0.f;
    vv[t] = y * inv;
    __syncthreads();
  }

  // deterministic sign: make largest-|v| component positive
  if (t == 0) {
    float bm = -1.f, bv = 1.f;
    for (int k = 0; k < COLS; ++k) {
      float a = fabsf(vv[k]);
      if (a > bm) { bm = a; bv = vv[k]; }
    }
    sgn_sh = (bv < 0.f) ? -1.f : 1.f;
  }
  __syncthreads();
  float sg = sgn_sh;

  // fold: out[r] = sum_k x[r,k]*w[k] - bias,  w = invstd*v*sg, bias = sum mean*w
  float wv = invstd[b * COLS + t] * vv[t] * sg;
  wvec[t] = wv;
  float pb = mean[b * COLS + t] * wv;
#pragma unroll
  for (int off = 16; off; off >>= 1) pb += __shfl_xor(pb, off, 32);
  if (lane == 0) part[wid] = pb;
  __syncthreads();
  float bias = part[0] + part[1] + part[2] + part[3] +
               part[4] + part[5] + part[6] + part[7];

  const float* xb = x + (size_t)b * ROWS * COLS;
#pragma unroll
  for (int rr = 0; rr < 3; ++rr) {
    int r = rr * COLS + t;
    const float4* row4 = (const float4*)(xb + (size_t)r * COLS);
    float acc = 0.f;
#pragma unroll 4
    for (int k4 = 0; k4 < COLS / 4; ++k4) {
      float4 xv = row4[k4];
      int k = k4 * 4;
      acc = fmaf(xv.x, wvec[k],     acc);
      acc = fmaf(xv.y, wvec[k + 1], acc);
      acc = fmaf(xv.z, wvec[k + 2], acc);
      acc = fmaf(xv.w, wvec[k + 3], acc);
    }
    out[(size_t)b * ROWS + r] = acc - bias;
  }
}

// ---------------------------------------------------------------------------
extern "C" void kernel_launch(void* const* d_in, const int* in_sizes, int n_in,
                              void* d_out, int out_size, void* d_ws, size_t ws_size,
                              hipStream_t stream) {
  (void)in_sizes; (void)n_in; (void)out_size; (void)ws_size;
  const float* x = (const float*)d_in[0];
  float* out = (float*)d_out;

  float* ws     = (float*)d_ws;
  float* mean   = ws;                       // 128*256 f32
  float* invstd = ws + BATCH * COLS;        // 128*256 f32
  float* G      = ws + 2 * BATCH * COLS;    // 128*256*256 f32 (32 MB)

  k_stats<<<BATCH, COLS, 0, stream>>>(x, mean, invstd);
  k_gram<<<dim3(4, BATCH, 1), 256, 0, stream>>>(x, mean, invstd, G);
  size_t smem = (size_t)COLS * 257 * sizeof(float);   // 257 KB < 320 KB/WGP
  k_power_proj<<<BATCH, COLS, smem, stream>>>(x, mean, invstd, G, out);
}
